// FP8Matmul_37357625541170
// MI455X (gfx1250) — 8-bit matrix operand order repaired and checked by reading, compiled, not run on hardware
//
#include <hip/hip_runtime.h>


#define M_DIM 8192
#define K_DIM 2048
#define N_DIM 4096

typedef int      v16i __attribute__((ext_vector_type(16)));
typedef float    v8f  __attribute__((ext_vector_type(8)));
typedef unsigned v4u  __attribute__((ext_vector_type(4)));
typedef unsigned v8u  __attribute__((ext_vector_type(8)));

__device__ __forceinline__ unsigned char f32_to_bf8(float x) {
    _Float16 h = (_Float16)x;
    unsigned short u = __builtin_bit_cast(unsigned short, h);
    if ((u & 0x7C00u) != 0x7C00u) {
        u = (unsigned short)(u + 0x7Fu + ((u >> 8) & 1u));
    } else if (u & 0x03FFu) {
        u |= 0x0300u;
    }
    return (unsigned char)(u >> 8);
}

__global__ __launch_bounds__(256)
void quant_bf8_rowmajor(const float* __restrict__ in,
                        unsigned char* __restrict__ out,
                        const float* __restrict__ scale_p,
                        long long n8) {
    long long i = (long long)blockIdx.x * 256 + threadIdx.x;
    if (i >= n8) return;
    float s = *scale_p;
    const float4* p = (const float4*)in + i * 2;
    float4 x0 = p[0];
    float4 x1 = p[1];
    unsigned long long r = 0;
    r |= (unsigned long long)f32_to_bf8(x0.x * s);
    r |= (unsigned long long)f32_to_bf8(x0.y * s) << 8;
    r |= (unsigned long long)f32_to_bf8(x0.z * s) << 16;
    r |= (unsigned long long)f32_to_bf8(x0.w * s) << 24;
    r |= (unsigned long long)f32_to_bf8(x1.x * s) << 32;
    r |= (unsigned long long)f32_to_bf8(x1.y * s) << 40;
    r |= (unsigned long long)f32_to_bf8(x1.z * s) << 48;
    r |= (unsigned long long)f32_to_bf8(x1.w * s) << 56;
    ((unsigned long long*)out)[i] = r;
}

__global__ __launch_bounds__(256)
void quant_bf8_transpose(const float* __restrict__ B,
                         unsigned char* __restrict__ Bt,
                         const float* __restrict__ scale_p) {
    __shared__ unsigned char lds[64][68];
    const int n0 = blockIdx.x * 64;
    const int k0 = blockIdx.y * 64;
    const float s = *scale_p;
    const int t  = threadIdx.x;
    const int kr = t >> 4;
    const int c  = t & 15;

    #pragma unroll
    for (int i = 0; i < 4; ++i) {
        int k = k0 + kr + i * 16;
        float4 x = *(const float4*)(B + (size_t)k * N_DIM + n0 + c * 4);
        unsigned int packed =
            (unsigned int)f32_to_bf8(x.x * s)        |
            ((unsigned int)f32_to_bf8(x.y * s) << 8)  |
            ((unsigned int)f32_to_bf8(x.z * s) << 16) |
            ((unsigned int)f32_to_bf8(x.w * s) << 24);
        *(unsigned int*)&lds[kr + i * 16][c * 4] = packed;
    }
    __syncthreads();

    const int n  = t >> 2;
    const int kq = (t & 3) * 16;
    union { uint4 q; unsigned char b[16]; } u;
    #pragma unroll
    for (int i = 0; i < 16; ++i) u.b[i] = lds[kq + i][n];
    *(uint4*)(Bt + (size_t)(n0 + n) * K_DIM + k0 + kq) = u.q;
}

__device__ __forceinline__ void tdm_load_tile_2d(unsigned ldsAddr,
                                                 const void* gaddr,
                                                 unsigned tensorD0,
                                                 unsigned tensorD1,
                                                 unsigned tileD0,
                                                 unsigned tileD1,
                                                 unsigned stride0) {
    const unsigned long long ga = (unsigned long long)gaddr;
    v4u g0;
    g0[0] = 1u;
    g0[1] = ldsAddr;
    g0[2] = (unsigned)ga;
    g0[3] = (unsigned)((ga >> 32) & 0x01FFFFFFu)
            | 0x80000000u;
    v8u g1;
    g1[0] = (1u << 20)
          | (4u << 22)
          | (3u << 25);
    g1[1] = (tensorD0 & 0xFFFFu) << 16;
    g1[2] = (tensorD0 >> 16) | ((tensorD1 & 0xFFFFu) << 16);
    g1[3] = (tensorD1 >> 16) | (tileD0 << 16);
    g1[4] = tileD1;
    g1[5] = stride0;
    g1[6] = 0;
    g1[7] = 0;
    asm volatile("tensor_load_to_lds %0, %1" :: "s"(g0), "s"(g1) : "memory");
}

#define LDS_ROW     144
#define TILE_BYTES  (128 * LDS_ROW)
#define STAGE_BYTES (2 * TILE_BYTES)

__global__ __launch_bounds__(256)
void bf8_wmma_gemm(const unsigned char* __restrict__ Aq,
                   const unsigned char* __restrict__ Bq,
                   float* __restrict__ C,
                   const float* __restrict__ s1p,
                   const float* __restrict__ s2p) {
    __shared__ unsigned char smem[2 * STAGE_BYTES];

    const int lane  = threadIdx.x & 31;
    const int wave  = threadIdx.x >> 5;
    const int waveM = wave & 3;
    const int waveN = wave >> 2;
    const int mBlock = blockIdx.y * 128;
    const int nBlock = blockIdx.x * 128;
    const int lo = lane & 15;
    const int hi = lane >> 4;

    const unsigned ldsBase = (unsigned)(unsigned long long)(const void*)smem;
    const unsigned char* aTileG = Aq + (size_t)mBlock * K_DIM;
    const unsigned char* bTileG = Bq + (size_t)nBlock * K_DIM;
    const bool issuer = (wave == 0);

    auto issue_stage = [&](int buf, int kk) {
        const unsigned dst = ldsBase + buf * STAGE_BYTES;
        tdm_load_tile_2d(dst,              aTileG + kk,
                         K_DIM, M_DIM - mBlock, 128, 128, K_DIM);
        tdm_load_tile_2d(dst + TILE_BYTES, bTileG + kk,
                         K_DIM, N_DIM - nBlock, 128, 128, K_DIM);
    };

    v8f acc[2][4];
    #pragma unroll
    for (int i = 0; i < 2; ++i)
        #pragma unroll
        for (int j = 0; j < 4; ++j) acc[i][j] = (v8f){};

    union AF { v16i v; unsigned long long u[8]; };
    union BF { v16i v; uint4 q[4]; };

    const int r0 = waveM * 32 + lo;
    const int NS = K_DIM / 128;

    if (issuer) issue_stage(0, 0);

    for (int s = 0; s < NS; ++s) {
        __builtin_amdgcn_s_wait_tensorcnt(0);
        __syncthreads();
        if (s + 1 < NS && issuer) issue_stage((s + 1) & 1, (s + 1) * 128);

        const unsigned char* aT = smem + (s & 1) * STAGE_BYTES;
        const unsigned char* bT = aT + TILE_BYTES;

        AF a0, a1;
        #pragma unroll
        for (int blk = 0; blk < 8; ++blk) {
            const int koff = (blk >> 1) * 32 + hi * 16 + (blk & 1) * 8;
            a0.u[blk] = *(const unsigned long long*)(aT + r0 * LDS_ROW + koff);
            a1.u[blk] = *(const unsigned long long*)(aT + (r0 + 16) * LDS_ROW + koff);
        }

        #pragma unroll
        for (int j = 0; j < 4; ++j) {
            const int col = waveN * 64 + j * 16 + lo;
            BF b;
            #pragma unroll
            for (int q = 0; q < 4; ++q) {
                const int koff = (q * 2 + hi) * 16;
                b.q[q] = *(const uint4*)(bT + col * LDS_ROW + koff);
            }
            acc[0][j] = __builtin_amdgcn_wmma_f32_16x16x128_bf8_bf8(
                a0.v, b.v, (short)0, acc[0][j], false, false);
            acc[1][j] = __builtin_amdgcn_wmma_f32_16x16x128_bf8_bf8(
                a1.v, b.v, (short)0, acc[1][j], false, false);
        }
    }

    const float inv = 1.0f / ((*s1p) * (*s2p));
    #pragma unroll
    for (int i = 0; i < 2; ++i)
        #pragma unroll
        for (int j = 0; j < 4; ++j) {
            const int col = nBlock + waveN * 64 + j * 16 + lo;
            #pragma unroll
            for (int r = 0; r < 8; ++r) {
                const int rowO = mBlock + waveM * 32 + i * 16 + hi * 8 + r;
                C[(size_t)rowO * N_DIM + col] = acc[i][j][r] * inv;
            }
        }
}

extern "C" void kernel_launch(void* const* d_in, const int* in_sizes, int n_in,
                              void* d_out, int out_size, void* d_ws, size_t ws_size,
                              hipStream_t stream) {
    const float* A  = (const float*)d_in[0];
    const float* B  = (const float*)d_in[1];
    const float* s1 = (const float*)d_in[2];
    const float* s2 = (const float*)d_in[3];
    float* out = (float*)d_out;

    unsigned char* Aq = (unsigned char*)d_ws;
    unsigned char* Bq = Aq + (size_t)M_DIM * K_DIM;

    const long long n8 = (long long)M_DIM * K_DIM / 8;
    quant_bf8_rowmajor<<<(unsigned)((n8 + 255) / 256), 256, 0, stream>>>(A, Aq, s1, n8);
    quant_bf8_transpose<<<dim3(N_DIM / 64, K_DIM / 64), 256, 0, stream>>>(B, Bq, s2);
    bf8_wmma_gemm<<<dim3(N_DIM / 128, M_DIM / 128), 256, 0, stream>>>(Aq, Bq, out, s1, s2);
}
